// AttentionHead_42159398977656
// MI455X (gfx1250) — compile-verified, matrix operand order checked by reading, not run on hardware
//
#include <hip/hip_runtime.h>
#include <hip/hip_bf16.h>

// ---------------------------------------------------------------------------
// Problem constants (from the reference)
// ---------------------------------------------------------------------------
#define BATCH 8
#define SEQ   2048
#define DIM   1024
#define HEAD  128
#define QT_TILES (SEQ / 16)           // 128 query/key 16-tiles per batch
static __device__ constexpr float SCALE = 0.08838834764831845f; // 1/sqrt(128)

// ---------------------------------------------------------------------------
// WMMA types (gfx1250, wave32): v_wmma_f32_16x16x32_bf16
// ---------------------------------------------------------------------------
typedef __attribute__((ext_vector_type(16))) __bf16 v16bf;
typedef __attribute__((ext_vector_type(8)))  __bf16 v8bf;
typedef __attribute__((ext_vector_type(8)))  float  v8f;

union V16 { v16bf v; v8bf h[2]; };

static __device__ inline v8f wmma_bf16(v16bf a, v16bf b, v8f c) {
  // 8 args: (neg_a, A, neg_b, B, c_mod, C, reuse_a, reuse_b)
  return __builtin_amdgcn_wmma_f32_16x16x32_bf16(
      false, a, false, b, (short)0, c, false, false);
}

// Async DMA of one 16-byte chunk: global -> LDS (tracked by ASYNCcnt).
// vdst = per-lane LDS byte address, vaddr = per-lane 64-bit global address.
static __device__ inline void async_ld16(void* lds_ptr, const void* gptr) {
  unsigned loff = (unsigned)(unsigned long)lds_ptr;   // flat LDS addr: low 32b = LDS offset
  asm volatile("global_load_async_to_lds_b128 %0, %1, off"
               :: "v"(loff), "v"(gptr) : "memory");
}
static __device__ inline void wait_async0() {
  asm volatile("s_wait_asynccnt 0x0" ::: "memory");
}

// ---------------------------------------------------------------------------
// Kernel 1: transpose + convert weights: W[d,h] f32 -> Wt[h,d] bf16
//   Makes WMMA B-operand tiles contiguous 16B loads along K(=d).
// ---------------------------------------------------------------------------
__global__ __launch_bounds__(256) void prep_wt_kernel(
    const float* __restrict__ Wq, const float* __restrict__ Wk,
    const float* __restrict__ Wv,
    __bf16* __restrict__ wtq, __bf16* __restrict__ wtk, __bf16* __restrict__ wtv)
{
  const int y = blockIdx.y;
  const float* W  = (y == 0) ? Wq : (y == 1) ? Wk : Wv;
  __bf16*      Wt = (y == 0) ? wtq : (y == 1) ? wtk : wtv;
  int idx = blockIdx.x * 256 + threadIdx.x;      // idx = d*HEAD + h, coalesced read
  if (idx < DIM * HEAD) {
    int d = idx / HEAD, h = idx % HEAD;
    Wt[(long)h * DIM + d] = (__bf16)W[idx];
  }
}

// ---------------------------------------------------------------------------
// Kernel 2: QKV projection.  grid = (128, 3), block = 256 (8 waves).
//   Wave w computes rows [ (bx*8+w)*16 , +16 ) x all 128 cols via 8x(D/32) WMMAs.
//   x is converted f32->bf16 on the fly (x lives in the 192MB L2 across passes).
//   y==0 -> Qb row-major bf16, y==1 -> Kb row-major bf16, y==2 -> Vt[B,H,S] bf16.
// ---------------------------------------------------------------------------
__global__ __launch_bounds__(256) void proj_kernel(
    const float* __restrict__ x,
    const __bf16* __restrict__ wtq, const __bf16* __restrict__ wtk,
    const __bf16* __restrict__ wtv,
    const float* __restrict__ bq, const float* __restrict__ bk,
    const float* __restrict__ bv,
    __bf16* __restrict__ Qb, __bf16* __restrict__ Kb, __bf16* __restrict__ Vt)
{
  const int y = blockIdx.y;
  const __bf16* Wt   = (y == 0) ? wtq : (y == 1) ? wtk : wtv;
  const float*  bias = (y == 0) ? bq  : (y == 1) ? bk  : bv;

  const int wave = threadIdx.x >> 5;
  const int lane = threadIdx.x & 31;
  const int ln16 = lane & 15;
  const int hi   = lane >> 4;                       // lane half: K-chunk select
  const long rowbase = (long)(blockIdx.x * 8 + wave) * 16;

  v8f acc[8];
#pragma unroll
  for (int n = 0; n < 8; ++n) acc[n] = (v8f)(0.f);

  const float* xrow = x + (rowbase + ln16) * DIM;   // A: lane = row M
  for (int d0 = 0; d0 < DIM; d0 += 32) {
    // A tile 16x32 bf16: lane half 0 holds K 0..7 & 16..23, half 1 holds 8..15 & 24..31
    V16 A;
    const int kb = d0 + hi * 8;
#pragma unroll
    for (int i = 0; i < 8; ++i) A.v[i]     = (__bf16)xrow[kb + i];
#pragma unroll
    for (int i = 0; i < 8; ++i) A.v[8 + i] = (__bf16)xrow[kb + 16 + i];
#pragma unroll
    for (int n = 0; n < 8; ++n) {
      const int h = n * 16 + ln16;                  // B: lane = col N
      const __bf16* wrow = Wt + (long)h * DIM + kb;
      V16 Bm;
      Bm.h[0] = *(const v8bf*)(wrow);
      Bm.h[1] = *(const v8bf*)(wrow + 16);
      acc[n] = wmma_bf16(A.v, Bm.v, acc[n]);
    }
  }

  // Epilogue: + bias, cvt bf16, scatter per C-layout (VGPR r -> M = r + hi*8)
#pragma unroll
  for (int n = 0; n < 8; ++n) {
    const int h = n * 16 + ln16;
    const float bb = bias[h];
#pragma unroll
    for (int r = 0; r < 8; ++r) {
      const long m = rowbase + r + hi * 8;
      const __bf16 o = (__bf16)(acc[n][r] + bb);
      if (y == 0)      Qb[m * HEAD + h] = o;
      else if (y == 1) Kb[m * HEAD + h] = o;
      else {
        const long bIdx = m >> 11, s = m & (SEQ - 1);
        Vt[(bIdx * HEAD + h) * (long)SEQ + s] = o;  // transposed for P@V B-operand
      }
    }
  }
}

// ---------------------------------------------------------------------------
// Kernel 3: per-key-column softmax stats over the QUERY axis (axis=1 softmax).
//   grid = (128 key tiles, B).  Wave w streams q-tiles kt+w, kt+w+8, ...
//   Online (m, z) per column; merge lane-halves via shfl_xor(16), waves via LDS.
//   Stores m[k] and 1/Z[k].
// ---------------------------------------------------------------------------
__global__ __launch_bounds__(256) void colstats_kernel(
    const __bf16* __restrict__ Qb, const __bf16* __restrict__ Kb,
    float* __restrict__ mbuf, float* __restrict__ rzbuf)
{
  __shared__ float sm[8][16];
  __shared__ float sz[8][16];
  const int kt = blockIdx.x, b = blockIdx.y;
  const int wave = threadIdx.x >> 5;
  const int lane = threadIdx.x & 31;
  const int ln16 = lane & 15;
  const int hi   = lane >> 4;
  const int key  = kt * 16 + ln16;

  // B operand (K^T tile) is fixed for the whole block: hoist the 4 K-chunks
  V16 Bk[4];
  const __bf16* krow = Kb + ((long)b * SEQ + key) * HEAD;
#pragma unroll
  for (int d = 0; d < 4; ++d) {
    Bk[d].h[0] = *(const v8bf*)(krow + d * 32 + hi * 8);
    Bk[d].h[1] = *(const v8bf*)(krow + d * 32 + hi * 8 + 16);
  }

  float mrun = -__builtin_inff(), zrun = 0.f;
  for (int qt = kt + wave; qt < QT_TILES; qt += 8) {
    const __bf16* qrow = Qb + ((long)b * SEQ + qt * 16 + ln16) * HEAD;
    v8f s = (v8f)(0.f);
#pragma unroll
    for (int d = 0; d < 4; ++d) {
      V16 A;
      A.h[0] = *(const v8bf*)(qrow + d * 32 + hi * 8);
      A.h[1] = *(const v8bf*)(qrow + d * 32 + hi * 8 + 16);
      s = wmma_bf16(A.v, Bk[d].v, s);
    }
    float vals[8], tmax = -__builtin_inff();
#pragma unroll
    for (int r = 0; r < 8; ++r) {
      const int q = qt * 16 + r + hi * 8;
      const float vv = (q >= key) ? s[r] * SCALE : -__builtin_inff();
      vals[r] = vv;
      tmax = fmaxf(tmax, vv);
    }
    const float mnew = fmaxf(mrun, tmax);
    if (mnew != -__builtin_inff()) {
      const float zs = (mrun == -__builtin_inff()) ? 0.f : __expf(mrun - mnew);
      float add = 0.f;
#pragma unroll
      for (int r = 0; r < 8; ++r)
        if (vals[r] != -__builtin_inff()) add += __expf(vals[r] - mnew);
      zrun = zrun * zs + add;
      mrun = mnew;
    }
  }

  // merge the two lane halves (same column N, rows M split across halves)
  const float mo = __shfl_xor(mrun, 16, 32);
  const float zo = __shfl_xor(zrun, 16, 32);
  const float mc = fmaxf(mrun, mo);
  float zc = 0.f;
  if (mc != -__builtin_inff()) {
    zc = ((mrun == -__builtin_inff()) ? 0.f : zrun * __expf(mrun - mc)) +
         ((mo   == -__builtin_inff()) ? 0.f : zo   * __expf(mo   - mc));
  }
  if (lane < 16) { sm[wave][lane] = mc; sz[wave][lane] = zc; }
  __syncthreads();

  if (threadIdx.x < 16) {
    float M = -__builtin_inff(), Z = 0.f;
#pragma unroll
    for (int w = 0; w < 8; ++w) {
      const float mw = sm[w][threadIdx.x], zw = sz[w][threadIdx.x];
      const float mn = fmaxf(M, mw);
      if (mn != -__builtin_inff()) {
        const float za = (M  == -__builtin_inff()) ? 0.f : Z  * __expf(M  - mn);
        const float zb = (mw == -__builtin_inff()) ? 0.f : zw * __expf(mw - mn);
        Z = za + zb;
        M = mn;
      }
    }
    const long o = (long)b * SEQ + kt * 16 + threadIdx.x;
    mbuf[o]  = M;
    rzbuf[o] = 1.0f / Z;   // diagonal element guarantees Z > 0
  }
}

// ---------------------------------------------------------------------------
// Kernel 4: O = P @ V, block-cooperative, 32 keys per step.
//   grid = (16, B), 8 waves; wave w owns q-tile qt = bx*8 + w (128 q rows/block).
//   K and V chunks are staged into LDS ONCE per block via the gfx1250 async
//   DMA path (global_load_async_to_lds_b128 / ASYNCcnt), double-buffered so the
//   next chunk's DMA overlaps this chunk's WMMAs.  Both WMMA B-operands are
//   then served by ds_load_b128 — 8x less global K/V traffic than per-wave.
//   P is bounced through a per-wave LDS slab (C-layout -> A-layout).
// ---------------------------------------------------------------------------
__global__ __launch_bounds__(256) void attn_out_kernel(
    const __bf16* __restrict__ Qb, const __bf16* __restrict__ Kb,
    const __bf16* __restrict__ Vt,
    const float* __restrict__ mbuf, const float* __restrict__ rzbuf,
    float* __restrict__ out)
{
  __shared__ __align__(16) __bf16 Ks[2][32][HEAD];     // 2 x 8 KB  [key][h]
  __shared__ __align__(16) __bf16 Vs[2][HEAD][32];     // 2 x 8 KB  [h][key]
  __shared__ __align__(16) __bf16 pshare[8][16][32];   // 8 KB [wave][q][key]

  const int b    = blockIdx.y;
  const int tid  = threadIdx.x;
  const int wave = tid >> 5;
  const int lane = tid & 31;
  const int ln16 = lane & 15;
  const int hi   = lane >> 4;
  const int qt   = blockIdx.x * 8 + wave;
  const int kend_w = qt * 16 + 16;             // this wave's causal key bound
  const int kmax   = (blockIdx.x + 1) * 128;   // block-wide key bound

  // Stage one 32-key chunk: K rows (32x128 bf16) + V columns (128x32 bf16).
  // 8 KB each => 512 x 16B chunks => 2 chunks per thread per tile.
  auto stage = [&](int buf, int kc) {
    const char* kg = (const char*)(Kb + ((long)b * SEQ + kc) * HEAD);
    char* ksl = (char*)&Ks[buf][0][0];
#pragma unroll
    for (int i = 0; i < 2; ++i) {
      const int o = (tid + i * 256) * 16;      // linear bytes; layout matches [32][128]
      async_ld16(ksl + o, kg + o);
    }
    char* vsl = (char*)&Vs[buf][0][0];
#pragma unroll
    for (int i = 0; i < 2; ++i) {
      const int t = tid + i * 256;
      const int h = t >> 2;                    // Vs row (64 B per row)
      const int cb = (t & 3) * 16;             // byte chunk within row
      const char* vg = (const char*)(Vt + ((long)b * HEAD + h) * SEQ + kc) + cb;
      async_ld16(vsl + t * 16, vg);
    }
  };

  // Hoist the Q A-operand (16 rows x 128 K) for this q-tile: 4 chunks of K=32
  V16 Aq[4];
  const __bf16* qrow = Qb + ((long)b * SEQ + qt * 16 + ln16) * HEAD;
#pragma unroll
  for (int d = 0; d < 4; ++d) {
    Aq[d].h[0] = *(const v8bf*)(qrow + d * 32 + hi * 8);
    Aq[d].h[1] = *(const v8bf*)(qrow + d * 32 + hi * 8 + 16);
  }

  v8f accO[8];
#pragma unroll
  for (int n = 0; n < 8; ++n) accO[n] = (v8f)(0.f);

  // prologue: stage chunk 0, publish
  stage(0, 0);
  wait_async0();
  __syncthreads();

  int buf = 0;
  for (int kc = 0; kc < kmax; kc += 32) {
    if (kc + 32 < kmax) stage(buf ^ 1, kc + 32);   // overlap DMA with compute

    if (kc < kend_w) {
      // --- scores for keys [kc, kc+16) and [kc+16, kc+32), B from LDS ---
      v8f s0 = (v8f)(0.f), s1 = (v8f)(0.f);
      const __bf16* k0l = &Ks[buf][ln16][0];
      const __bf16* k1l = &Ks[buf][16 + ln16][0];
#pragma unroll
      for (int d = 0; d < 4; ++d) {
        V16 B0, B1;
        B0.h[0] = *(const v8bf*)(k0l + d * 32 + hi * 8);
        B0.h[1] = *(const v8bf*)(k0l + d * 32 + hi * 8 + 16);
        s0 = wmma_bf16(Aq[d].v, B0.v, s0);
        B1.h[0] = *(const v8bf*)(k1l + d * 32 + hi * 8);
        B1.h[1] = *(const v8bf*)(k1l + d * 32 + hi * 8 + 16);
        s1 = wmma_bf16(Aq[d].v, B1.v, s1);
      }
      // --- P[q,k] = exp(s/sqrt(d) - m[k]) / Z[k], causal-masked ---
      const int key0 = kc + ln16, key1 = key0 + 16;
      const float m0 = mbuf[(long)b * SEQ + key0], r0 = rzbuf[(long)b * SEQ + key0];
      const float m1 = mbuf[(long)b * SEQ + key1], r1 = rzbuf[(long)b * SEQ + key1];
#pragma unroll
      for (int r = 0; r < 8; ++r) {
        const int q  = qt * 16 + r + hi * 8;
        const float p0 = (q >= key0) ? __expf(s0[r] * SCALE - m0) * r0 : 0.f;
        const float p1 = (q >= key1) ? __expf(s1[r] * SCALE - m1) * r1 : 0.f;
        pshare[wave][r + hi * 8][ln16]      = (__bf16)p0;   // C-layout scatter
        pshare[wave][r + hi * 8][16 + ln16] = (__bf16)p1;
      }
      // per-wave slab, wave-synchronous: drain DS pipe before re-reading
      asm volatile("s_wait_dscnt 0x0" ::: "memory");

      // reload P in WMMA-A layout (lane = row q, K chunks by lane half)
      V16 Ap;
      const __bf16* prow = &pshare[wave][ln16][0];
      Ap.h[0] = *(const v8bf*)(prow + hi * 8);
      Ap.h[1] = *(const v8bf*)(prow + hi * 8 + 16);

      // --- O += P @ V over the 32-key chunk, B from LDS ---
#pragma unroll
      for (int n = 0; n < 8; ++n) {
        const __bf16* vl = &Vs[buf][n * 16 + ln16][0] + hi * 8;
        V16 Bv;
        Bv.h[0] = *(const v8bf*)(vl);
        Bv.h[1] = *(const v8bf*)(vl + 16);
        accO[n] = wmma_bf16(Ap.v, Bv.v, accO[n]);
      }
    }

    // publish next buffer / protect current one from early overwrite
    wait_async0();
    __syncthreads();
    buf ^= 1;
  }

  // store O tile (f32, row-major [B,S,H])
#pragma unroll
  for (int n = 0; n < 8; ++n) {
    const int h = n * 16 + ln16;
#pragma unroll
    for (int r = 0; r < 8; ++r) {
      const long q = qt * 16 + r + hi * 8;
      out[((long)b * SEQ + q) * HEAD + h] = accO[n][r];
    }
  }
}

// ---------------------------------------------------------------------------
// Host-side orchestration
// ---------------------------------------------------------------------------
extern "C" void kernel_launch(void* const* d_in, const int* in_sizes, int n_in,
                              void* d_out, int out_size, void* d_ws, size_t ws_size,
                              hipStream_t stream) {
  const float* x  = (const float*)d_in[0];
  const float* Wq = (const float*)d_in[1];
  const float* bq = (const float*)d_in[2];
  const float* Wk = (const float*)d_in[3];
  const float* bk = (const float*)d_in[4];
  const float* Wv = (const float*)d_in[5];
  const float* bv = (const float*)d_in[6];
  float* out = (float*)d_out;

  char* ws = (char*)d_ws;
  const size_t WT_BYTES = (size_t)DIM * HEAD * sizeof(__bf16);         // 256 KB
  const size_t QB_BYTES = (size_t)BATCH * SEQ * HEAD * sizeof(__bf16); // 4 MB
  const size_t ST_BYTES = (size_t)BATCH * SEQ * sizeof(float);         // 64 KB

  size_t off = 0;
  __bf16* wtq = (__bf16*)(ws + off); off += WT_BYTES;
  __bf16* wtk = (__bf16*)(ws + off); off += WT_BYTES;
  __bf16* wtv = (__bf16*)(ws + off); off += WT_BYTES;
  __bf16* Qb  = (__bf16*)(ws + off); off += QB_BYTES;
  __bf16* Kb  = (__bf16*)(ws + off); off += QB_BYTES;
  __bf16* Vt  = (__bf16*)(ws + off); off += QB_BYTES;
  float*  mb  = (float*)(ws + off);  off += ST_BYTES;
  float*  rz  = (float*)(ws + off);  off += ST_BYTES;
  // total ~13.5 MB of workspace

  // 1) weight transpose+convert (f32 -> bf16, [D,H] -> [H,D])
  prep_wt_kernel<<<dim3((DIM * HEAD) / 256, 3), 256, 0, stream>>>(
      Wq, Wk, Wv, wtq, wtk, wtv);

  // 2) QKV projections via WMMA (x converted on the fly; x stays hot in L2)
  proj_kernel<<<dim3((BATCH * SEQ) / 128, 3), 256, 0, stream>>>(
      x, wtq, wtk, wtv, bq, bk, bv, Qb, Kb, Vt);

  // 3) per-key-column softmax stats (softmax over the query axis, causal)
  colstats_kernel<<<dim3(QT_TILES, BATCH), 256, 0, stream>>>(Qb, Kb, mb, rz);

  // 4) O = normalized-P @ V with async-LDS double-buffered K/V staging
  attn_out_kernel<<<dim3(QT_TILES / 8, BATCH), 256, 0, stream>>>(
      Qb, Kb, Vt, mb, rz, out);
}
